// CausalSelfAttention_13443247637319
// MI455X (gfx1250) — matrix operand order repaired and checked by reading, not compiled, not run on hardware
//
#include <hip/hip_runtime.h>
#include <hip/hip_bf16.h>

typedef __attribute__((ext_vector_type(16))) __bf16         v16bf;
typedef __attribute__((ext_vector_type(8)))  float          v8f;
typedef __attribute__((ext_vector_type(8)))  unsigned short ushort8;
typedef __attribute__((ext_vector_type(16))) unsigned short ushort16;
typedef __attribute__((ext_vector_type(4)))  unsigned int   uint32x4;
typedef __attribute__((ext_vector_type(8)))  int            int32x8;
typedef __attribute__((ext_vector_type(4)))  int            int32x4;

#define B_   2
#define S_   2048
#define E_   1024
#define H_   16
#define D_   64
#define NROW (B_ * S_)   /* 4096 */
#define EE   (E_ * E_)   /* 1048576 = 2^20 */
#define LDK  72          /* padded LDS row (elems): 144B rows, 16B aligned, conflict-free */

#if defined(__has_builtin)
#  if __has_builtin(__builtin_amdgcn_tensor_load_to_lds)
#    define HAVE_TDM 1
#  endif
#endif
#ifndef HAVE_TDM
#  define HAVE_TDM 0
#endif

__device__ __forceinline__ unsigned short f32_to_bf16_rne(float f) {
  unsigned int u = __builtin_bit_cast(unsigned int, f);
  u += 0x7FFFu + ((u >> 16) & 1u);
  return (unsigned short)(u >> 16);
}

__device__ __forceinline__ v16bf pack16(ushort8 a, ushort8 b) {
  ushort16 t = __builtin_shufflevector(a, b, 0,1,2,3,4,5,6,7,8,9,10,11,12,13,14,15);
  return __builtin_bit_cast(v16bf, t);
}

__device__ __forceinline__ v8f v8f_zero() {
  v8f z;
#pragma unroll
  for (int i = 0; i < 8; ++i) z[i] = 0.f;
  return z;
}

// A fragment (16x32 bf16): lanes 0-15 row=lane, K [0..7],[16..23]; lanes 16-31 K [8..15],[24..31]
__device__ __forceinline__ v16bf load_fragA(const unsigned short* p0, int ld, int lane) {
  const int r  = lane & 15;
  const int ko = (lane & 16) ? 8 : 0;
  const unsigned short* p = p0 + r * ld + ko;
  ushort8 a = *(const ushort8*)(p);
  ushort8 b = *(const ushort8*)(p + 16);
  return pack16(a, b);
}

// B fragment (32x16 bf16, NT: column n = memory row n): lanes 0-15 K [0..7],[16..23]; lanes 16-31 K [8..15],[24..31]
__device__ __forceinline__ v16bf load_fragB(const unsigned short* p0, int ld, int lane) {
  const int n  = lane & 15;
  const int kh = (lane & 16) ? 8 : 0;
  const unsigned short* p = p0 + n * ld + kh;
  ushort8 a = *(const ushort8*)(p);
  ushort8 b = *(const ushort8*)(p + 16);
  return pack16(a, b);
}

__device__ __forceinline__ v8f wmma_bf16(v16bf a, v16bf b, v8f c) {
  return __builtin_amdgcn_wmma_f32_16x16x32_bf16(false, a, false, b, (short)0, c,
                                                 false, false);
}

#if HAVE_TDM
// 2D bf16 tile load 64 rows x 64 elems -> LDS with +4 DWORD pad per 32-DWORD row (ld=72)
__device__ __forceinline__ void tdm_load_tile(unsigned lds_off, const unsigned short* gsrc,
                                              unsigned tensor_d0, unsigned tensor_d1,
                                              unsigned stride0) {
  unsigned long long ga = (unsigned long long)(size_t)(const void*)gsrc;
  uint32x4 g0;
  g0[0] = 1u;                                      // count=1, user descriptor
  g0[1] = lds_off;                                 // lds_addr (bytes)
  g0[2] = (unsigned)(ga & 0xFFFFFFFFu);            // global_addr[31:0]
  g0[3] = (unsigned)((ga >> 32) & 0x01FFFFFFu) | (2u << 30);  // addr[56:32], type=2
  int32x8 g1;
  // data_size=1(2B) | pad_enable | pad_interval=4(32 DW) | pad_amount=3(4 DW)
  g1[0] = (int)0x07110000u;
  g1[1] = (int)((tensor_d0 & 0xFFFFu) << 16);                       // tensor_dim0 lo
  g1[2] = (int)(((tensor_d0 >> 16) & 0xFFFFu) | ((tensor_d1 & 0xFFFFu) << 16));
  g1[3] = (int)(((tensor_d1 >> 16) & 0xFFFFu) | (64u << 16));       // tile_dim0 = 64
  g1[4] = (int)64;                                                  // tile_dim1 = 64
  g1[5] = (int)stride0;                                             // dim0 stride lo
  g1[6] = 0;
  g1[7] = 0;
  int32x4 z4; z4[0] = z4[1] = z4[2] = z4[3] = 0;
#if defined(__clang_major__) && (__clang_major__ >= 23)
  int32x8 z8;
#pragma unroll
  for (int i = 0; i < 8; ++i) z8[i] = 0;
  __builtin_amdgcn_tensor_load_to_lds(g0, g1, z4, z4, z8, 0);
#else
  __builtin_amdgcn_tensor_load_to_lds(g0, g1, z4, z4, 0);
#endif
}
#endif

// ---------------- Kernel 1: fp32 -> bf16 conversion --------------------------
__global__ void cvt_bf16_kernel(const float* __restrict__ hs,
                                const float* __restrict__ wq,
                                const float* __restrict__ wk,
                                const float* __restrict__ wv,
                                unsigned short* __restrict__ hid,
                                unsigned short* __restrict__ w) {
  const int tot = NROW * E_ + 3 * EE;
  for (int i = blockIdx.x * blockDim.x + threadIdx.x; i < tot;
       i += gridDim.x * blockDim.x) {
    if (i < NROW * E_) {
      hid[i] = f32_to_bf16_rne(hs[i]);
    } else {
      int j = i - NROW * E_;
      int which = j >> 20;
      int off = j & (EE - 1);
      const float* src = (which == 0) ? wq : (which == 1) ? wk : wv;
      w[j] = f32_to_bf16_rne(src[off]);
    }
  }
}

// ---------------- Kernel 2: QKV projection (ping-pong NT WMMA GEMM) ----------
// grid (32, 16, 3), block 256 (8 waves). Wave: 16 rows x 64 cols, K=1024.
// K loop unrolled x2 with two named fragment buffers: each buffer is filled and
// consumed in alternate half-iterations, so no v_mov rotation copies and the
// global_load latency of one half overlaps the WMMAs of the other.
__global__ void __launch_bounds__(256, 1)
qkv_proj_kernel(const unsigned short* __restrict__ hid,
                const unsigned short* __restrict__ w,
                const float* __restrict__ bq,
                const float* __restrict__ bk,
                const float* __restrict__ bv,
                unsigned short* __restrict__ Qh,
                unsigned short* __restrict__ Kh,
                unsigned short* __restrict__ Vt) {
  const int z    = blockIdx.z;
  const int lane = threadIdx.x & 31;
  const int wave = threadIdx.x >> 5;
  const int m0   = blockIdx.x * 128 + wave * 16;
  const int n0   = blockIdx.y * 64;
  const unsigned short* wz = w + (size_t)z * EE;
  const float* bias = (z == 0) ? bq : (z == 1) ? bk : bv;

  const unsigned short* arow  = hid + (size_t)m0 * E_;
  const unsigned short* wrow0 = wz + (size_t)(n0 +  0) * E_;
  const unsigned short* wrow1 = wz + (size_t)(n0 + 16) * E_;
  const unsigned short* wrow2 = wz + (size_t)(n0 + 32) * E_;
  const unsigned short* wrow3 = wz + (size_t)(n0 + 48) * E_;

  v8f acc[4];
#pragma unroll
  for (int j = 0; j < 4; ++j) acc[j] = v8f_zero();

  // buffer 0 primed with k = 0
  v16bf a0 = load_fragA(arow, E_, lane);
  v16bf b00 = load_fragB(wrow0, E_, lane);
  v16bf b01 = load_fragB(wrow1, E_, lane);
  v16bf b02 = load_fragB(wrow2, E_, lane);
  v16bf b03 = load_fragB(wrow3, E_, lane);
  v16bf a1, b10, b11, b12, b13;

  for (int kb = 0; kb < E_ - 64; kb += 64) {
    // fill buffer 1 with k = kb+32, consume buffer 0 (k = kb)
    a1  = load_fragA(arow  + kb + 32, E_, lane);
    b10 = load_fragB(wrow0 + kb + 32, E_, lane);
    b11 = load_fragB(wrow1 + kb + 32, E_, lane);
    b12 = load_fragB(wrow2 + kb + 32, E_, lane);
    b13 = load_fragB(wrow3 + kb + 32, E_, lane);
    acc[0] = wmma_bf16(a0, b00, acc[0]);
    acc[1] = wmma_bf16(a0, b01, acc[1]);
    acc[2] = wmma_bf16(a0, b02, acc[2]);
    acc[3] = wmma_bf16(a0, b03, acc[3]);
    // fill buffer 0 with k = kb+64, consume buffer 1 (k = kb+32)
    a0  = load_fragA(arow  + kb + 64, E_, lane);
    b00 = load_fragB(wrow0 + kb + 64, E_, lane);
    b01 = load_fragB(wrow1 + kb + 64, E_, lane);
    b02 = load_fragB(wrow2 + kb + 64, E_, lane);
    b03 = load_fragB(wrow3 + kb + 64, E_, lane);
    acc[0] = wmma_bf16(a1, b10, acc[0]);
    acc[1] = wmma_bf16(a1, b11, acc[1]);
    acc[2] = wmma_bf16(a1, b12, acc[2]);
    acc[3] = wmma_bf16(a1, b13, acc[3]);
  }
  // tail: k = E_-64 (in buffer 0) and k = E_-32 (load into buffer 1)
  a1  = load_fragA(arow  + E_ - 32, E_, lane);
  b10 = load_fragB(wrow0 + E_ - 32, E_, lane);
  b11 = load_fragB(wrow1 + E_ - 32, E_, lane);
  b12 = load_fragB(wrow2 + E_ - 32, E_, lane);
  b13 = load_fragB(wrow3 + E_ - 32, E_, lane);
  acc[0] = wmma_bf16(a0, b00, acc[0]);
  acc[1] = wmma_bf16(a0, b01, acc[1]);
  acc[2] = wmma_bf16(a0, b02, acc[2]);
  acc[3] = wmma_bf16(a0, b03, acc[3]);
  acc[0] = wmma_bf16(a1, b10, acc[0]);
  acc[1] = wmma_bf16(a1, b11, acc[1]);
  acc[2] = wmma_bf16(a1, b12, acc[2]);
  acc[3] = wmma_bf16(a1, b13, acc[3]);

  const int hi8 = (lane & 16) ? 8 : 0;
  const int nl  = lane & 15;
#pragma unroll
  for (int j = 0; j < 4; ++j) {
    const int n = n0 + j * 16 + nl;
    const int h = n >> 6, d = n & 63;
    const float bb = bias[n];
#pragma unroll
    for (int r = 0; r < 8; ++r) {
      const int m = m0 + r + hi8;
      const int b = m >> 11, t = m & (S_ - 1);
      const int bh = b * H_ + h;
      unsigned short val = f32_to_bf16_rne(acc[j][r] + bb);
      if (z == 2)      Vt[((size_t)bh * D_ + d) * S_ + t] = val;  // V transposed [d][t]
      else if (z == 0) Qh[((size_t)bh * S_ + t) * D_ + d] = val;
      else             Kh[((size_t)bh * S_ + t) * D_ + d] = val;
    }
  }
}

// ---------------- Kernel 3: causal flash attention ---------------------------
// grid (S/64, B*H), block 128 (4 waves). Wave owns 16 query rows.
// K/V 64x64 tiles staged in LDS once per block (TDM double-buffered prefetch).
__global__ void __launch_bounds__(128, 1)
flash_attn_kernel(const unsigned short* __restrict__ Qh,
                  const unsigned short* __restrict__ Kh,
                  const unsigned short* __restrict__ Vt,
                  const float* __restrict__ amask,
                  float* __restrict__ out) {
  const int tid  = threadIdx.x;
  const int lane = tid & 31;
  const int wave = tid >> 5;                  // 0..3
  const int bh   = blockIdx.y;                // 0..31
  const int b    = bh >> 4, h = bh & 15;
  const int qblk = blockIdx.x * 64;
  const int q0   = qblk + wave * 16;
  const int hi8  = (lane & 16) ? 8 : 0;
  const int nl   = lane & 15;

  const unsigned short* Qp = Qh + (size_t)bh * S_ * D_;
  const unsigned short* Kp = Kh + (size_t)bh * S_ * D_;
  const unsigned short* Vp = Vt + (size_t)bh * D_ * S_;
  const float* am = amask + (size_t)b * S_;

  __shared__ unsigned short KT[2][64 * LDK];
  __shared__ unsigned short VT[2][64 * LDK];
  __shared__ unsigned short Pl[4][16 * 80];

  const v16bf qf0 = load_fragA(Qp + (size_t)q0 * D_ + 0,  D_, lane);
  const v16bf qf1 = load_fragA(Qp + (size_t)q0 * D_ + 32, D_, lane);

  float m_i[8], l_i[8];
  v8f o[4];
#pragma unroll
  for (int r = 0; r < 8; ++r) { m_i[r] = -__builtin_inff(); l_i[r] = 0.f; }
#pragma unroll
  for (int j = 0; j < 4; ++j) o[j] = v8f_zero();

  const float scale = 0.125f;                 // 1/sqrt(64)
  const int nblk = (qblk >> 6) + 1;

#if HAVE_TDM
  if (wave == 0) {
    tdm_load_tile((unsigned)(size_t)(void*)&KT[0][0], Kp, 64u, (unsigned)S_, 64u);
    tdm_load_tile((unsigned)(size_t)(void*)&VT[0][0], Vp, (unsigned)S_, 64u, (unsigned)S_);
  }
#endif

  for (int it = 0; it < nblk; ++it) {
    const int k0 = it << 6;
#if HAVE_TDM
    const int buf = it & 1;
    if (wave == 0) {
      if (it + 1 < nblk) {
        const int kn = (it + 1) << 6;
        tdm_load_tile((unsigned)(size_t)(void*)&KT[buf ^ 1][0],
                      Kp + (size_t)kn * D_, 64u, (unsigned)(S_ - kn), 64u);
        tdm_load_tile((unsigned)(size_t)(void*)&VT[buf ^ 1][0],
                      Vp + kn, (unsigned)S_, 64u, (unsigned)S_);
        __builtin_amdgcn_s_wait_tensorcnt((unsigned short)2);
      } else {
        __builtin_amdgcn_s_wait_tensorcnt((unsigned short)0);
      }
    }
    __syncthreads();
#else
    const int buf = 0;
    __syncthreads();
    // cooperative staging: 64 rows x 64 elems, 8-elem chunks, 512 chunks / 128 threads
    for (int i = tid; i < 512; i += 128) {
      const int rr = i >> 3, cc = (i & 7) * 8;
      *(ushort8*)(&KT[0][rr * LDK + cc]) =
          *(const ushort8*)(Kp + (size_t)(k0 + rr) * D_ + cc);
      *(ushort8*)(&VT[0][rr * LDK + cc]) =
          *(const ushort8*)(Vp + (size_t)rr * S_ + k0 + cc);
    }
    __syncthreads();
#endif

    // ---- S = Q K^T over a 16x64 strip (8 WMMAs, B fragments from LDS) ----
    v8f s[4];
#pragma unroll
    for (int c = 0; c < 4; ++c) {
      v16bf kf0 = load_fragB(&KT[buf][(c * 16) * LDK + 0],  LDK, lane);
      v16bf kf1 = load_fragB(&KT[buf][(c * 16) * LDK + 32], LDK, lane);
      v8f t = v8f_zero();
      t = wmma_bf16(qf0, kf0, t);
      t = wmma_bf16(qf1, kf1, t);
      s[c] = t;
    }

    const bool diag = (k0 == qblk);
    float lm[8];
#pragma unroll
    for (int r = 0; r < 8; ++r) lm[r] = -__builtin_inff();
#pragma unroll
    for (int c = 0; c < 4; ++c) {
      const int col = k0 + c * 16 + nl;
      const float a = am[col];
#pragma unroll
      for (int r = 0; r < 8; ++r) {
        float v = s[c][r] * scale + a;
        if (diag && (col > q0 + r + hi8)) v = -__builtin_inff();
        s[c][r] = v;
        lm[r] = fmaxf(lm[r], v);
      }
    }
#pragma unroll
    for (int r = 0; r < 8; ++r)
      for (int off = 1; off < 16; off <<= 1)
        lm[r] = fmaxf(lm[r], __shfl_xor(lm[r], off, 32));

    float corr[8], rs[8];
#pragma unroll
    for (int r = 0; r < 8; ++r) {
      const float mn = fmaxf(m_i[r], lm[r]);
      corr[r] = __expf(m_i[r] - mn);
      m_i[r]  = mn;
      rs[r]   = 0.f;
    }
#pragma unroll
    for (int c = 0; c < 4; ++c) {
#pragma unroll
      for (int r = 0; r < 8; ++r) {
        const float p = __expf(s[c][r] - m_i[r]);
        rs[r] += p;
        Pl[wave][(r + hi8) * 80 + c * 16 + nl] = f32_to_bf16_rne(p);
      }
    }
#pragma unroll
    for (int r = 0; r < 8; ++r) {
      for (int off = 1; off < 16; off <<= 1)
        rs[r] += __shfl_xor(rs[r], off, 32);
      l_i[r] = l_i[r] * corr[r] + rs[r];
    }
#pragma unroll
    for (int j = 0; j < 4; ++j)
#pragma unroll
      for (int r = 0; r < 8; ++r) o[j][r] *= corr[r];

    asm volatile("s_wait_dscnt 0" ::: "memory");   // P stores visible before re-read

    const v16bf pf0 = load_fragA(&Pl[wave][0],      80, lane);
    const v16bf pf1 = load_fragA(&Pl[wave][0] + 32, 80, lane);
#pragma unroll
    for (int j = 0; j < 4; ++j) {
      v16bf vf0 = load_fragB(&VT[buf][(j * 16) * LDK + 0],  LDK, lane);
      v16bf vf1 = load_fragB(&VT[buf][(j * 16) * LDK + 32], LDK, lane);
      o[j] = wmma_bf16(pf0, vf0, o[j]);
      o[j] = wmma_bf16(pf1, vf1, o[j]);
    }
#if HAVE_TDM
    __syncthreads();   // all waves done with buf before it is re-filled
#endif
  }

  // ---- epilogue: normalize and write [b, t, h*64 + d] ----
#pragma unroll
  for (int r = 0; r < 8; ++r) {
    const int t = q0 + r + hi8;
    const float inv = 1.f / l_i[r];
#pragma unroll
    for (int j = 0; j < 4; ++j)
      out[((size_t)(b * S_ + t)) * E_ + h * D_ + j * 16 + nl] = o[j][r] * inv;
  }
}

// -----------------------------------------------------------------------------
extern "C" void kernel_launch(void* const* d_in, const int* in_sizes, int n_in,
                              void* d_out, int out_size, void* d_ws, size_t ws_size,
                              hipStream_t stream) {
  (void)in_sizes; (void)n_in; (void)out_size; (void)ws_size;
  const float* hs = (const float*)d_in[0];
  const float* amask = (const float*)d_in[1];
  const float* Wq = (const float*)d_in[2];
  const float* bq = (const float*)d_in[3];
  const float* Wk = (const float*)d_in[4];
  const float* bk = (const float*)d_in[5];
  const float* Wv = (const float*)d_in[6];
  const float* bv = (const float*)d_in[7];
  float* out = (float*)d_out;

  unsigned short* ws  = (unsigned short*)d_ws;
  unsigned short* hid = ws;                              // 4 Mi bf16
  unsigned short* w   = hid + (size_t)NROW * E_;         // 3 Mi bf16
  unsigned short* Qh  = w + 3 * (size_t)EE;              // 4 Mi bf16
  unsigned short* Kh  = Qh + (size_t)NROW * E_;          // 4 Mi bf16
  unsigned short* Vt  = Kh + (size_t)NROW * E_;          // 4 Mi bf16  (~38 MB total)

  cvt_bf16_kernel<<<dim3(1792), dim3(256), 0, stream>>>(hs, Wq, Wk, Wv, hid, w);
  qkv_proj_kernel<<<dim3(32, 16, 3), dim3(256), 0, stream>>>(hid, w, bq, bk, bv,
                                                             Qh, Kh, Vt);
  flash_attn_kernel<<<dim3(32, 32), dim3(128), 0, stream>>>(Qh, Kh, Vt, amask, out);
}
